// SDM_GNN_41412074668241
// MI455X (gfx1250) — matrix operand order repaired and checked by reading, not compiled, not run on hardware
//
#include <hip/hip_runtime.h>
#include <hip/hip_bf16.h>
#include <math.h>

typedef _Float16 h16;
typedef __attribute__((ext_vector_type(16))) _Float16 v16h;
typedef __attribute__((ext_vector_type(8)))  _Float16 v8h;
typedef __attribute__((ext_vector_type(4)))  _Float16 v4h;
typedef __attribute__((ext_vector_type(8)))  float    v8f;

#define B_ROWS 65536
#define DEG    32

// ---------------------------------------------------------------------------
// Fragment loaders (wave32 WMMA; this code's contiguous k = 16*(lane/16) + i in both A and B)
// ---------------------------------------------------------------------------

// A fragment: 16x32 f16 tile from row-major f16 matrix (lda in halfs).
// lanes 0-15: row M=lane, K = k0+0..15
// lanes16-31: row M=lane-16, K = k0+16..31
__device__ __forceinline__ v16h load_a_frag(const h16* __restrict__ Ab, int lda,
                                            int k0, int lane) {
  int row = lane & 15;
  int off = (lane >> 4) << 4;            // 0 or 16
  const h16* p = Ab + (size_t)row * lda + k0 + off;
  v8h lo = *(const v8h*)(p);
  v8h hi = *(const v8h*)(p + 8);
  v16h r;
#pragma unroll
  for (int i = 0; i < 8; ++i) { r[i] = lo[i]; r[i + 8] = hi[i]; }
  return r;
}

// B fragment: 32x16 f16 tile from the pre-swizzled weight buffer.
// Swizzle (done by swz_kernel): element (k,n) of a K x 128 weight lives at
//   (((k/32)*8 + n/16)*32 + ((k&16)|(n&15)))*16 + (k&15)
// so each lane reads its 16 halfs (K = kb..kb+15 for its column) contiguously.
__device__ __forceinline__ v16h load_b_frag(const h16* __restrict__ W,
                                            int kt, int nt, int lane) {
  const h16* p = W + (size_t)(((kt * 8 + nt) * 32 + lane) * 16);
  v8h lo = *(const v8h*)(p);
  v8h hi = *(const v8h*)(p + 8);
  v16h r;
#pragma unroll
  for (int i = 0; i < 8; ++i) { r[i] = lo[i]; r[i + 8] = hi[i]; }
  return r;
}

__device__ __forceinline__ v4h f4_to_h4(float x, float y, float z, float w) {
  v4h h; h[0] = (h16)x; h[1] = (h16)y; h[2] = (h16)z; h[3] = (h16)w; return h;
}

// ---------------------------------------------------------------------------
// Weight convert + B-fragment swizzle (f32 [K][128] -> f16 swizzled)
// ---------------------------------------------------------------------------
__global__ void swz_kernel(const float* __restrict__ w, h16* __restrict__ dst, int K) {
  int tid = blockIdx.x * 256 + threadIdx.x;
  if (tid >= K * 128) return;
  int k = tid >> 7, n = tid & 127;
  int kt = k >> 5, kk = k & 31, nt = n >> 4, nn = n & 15;
  int di = (((kt * 8 + nt) * 32) + ((kk & 16) | nn)) * 16 + (kk & 15);
  dst[di] = (h16)w[(size_t)k * 128 + n];
}

// ---------------------------------------------------------------------------
// Gather + neighbor-mean encoder front end.  One wave per batch row.
// Each lane owns 4 consecutive feature columns (float4): full 512B row per wave.
// ---------------------------------------------------------------------------
__global__ __launch_bounds__(256) void agg_kernel(
    const int* __restrict__ bn, const float* __restrict__ feat,
    const int* __restrict__ n0, const int* __restrict__ n1,
    const float* __restrict__ motif,
    h16* __restrict__ Xc,    // [B][384] f16 : self | mean0 | mean1
    h16* __restrict__ Xfc,   // [B][256] f16 : cols 128..255 <- motif
    float* __restrict__ Zm)  // [B][128] f32 : motif copy for fusion
{
  int row  = blockIdx.x * 8 + (threadIdx.x >> 5);
  int lane = threadIdx.x & 31;
  int node = bn[row];

  float4 s = ((const float4*)(feat + (size_t)node * 128))[lane];
  *(v4h*)(Xc + (size_t)row * 384 + lane * 4) = f4_to_h4(s.x, s.y, s.z, s.w);

  const float sc = 1.0f / 32.0f;
  {
    int nb = n0[(size_t)node * DEG + lane];
    float ax = 0.f, ay = 0.f, az = 0.f, aw = 0.f;
    for (int d = 0; d < DEG; ++d) {
      int idx = __shfl(nb, d, 32);
      float4 f = ((const float4*)(feat + (size_t)idx * 128))[lane];
      ax += f.x; ay += f.y; az += f.z; aw += f.w;
    }
    *(v4h*)(Xc + (size_t)row * 384 + 128 + lane * 4) =
        f4_to_h4(ax * sc, ay * sc, az * sc, aw * sc);
  }
  {
    int nb = n1[(size_t)node * DEG + lane];
    float ax = 0.f, ay = 0.f, az = 0.f, aw = 0.f;
    for (int d = 0; d < DEG; ++d) {
      int idx = __shfl(nb, d, 32);
      float4 f = ((const float4*)(feat + (size_t)idx * 128))[lane];
      ax += f.x; ay += f.y; az += f.z; aw += f.w;
    }
    *(v4h*)(Xc + (size_t)row * 384 + 256 + lane * 4) =
        f4_to_h4(ax * sc, ay * sc, az * sc, aw * sc);
  }

  float4 m = ((const float4*)(motif + (size_t)node * 128))[lane];
  ((float4*)(Zm + (size_t)row * 128))[lane] = m;
  *(v4h*)(Xfc + (size_t)row * 256 + 128 + lane * 4) = f4_to_h4(m.x, m.y, m.z, m.w);
}

// ---------------------------------------------------------------------------
// WMMA GEMM: [B,KDIM](f16) @ [KDIM,128](f16,swizzled) + bias, epilogue.
// Block = 8 waves; block owns one 16-row M tile; wave w owns N tile w.
// EPI: 0 = tanh->f16 | 1 = id->f32 AND f16 | 2 = relu->f16 | 3 = id->f32
// ---------------------------------------------------------------------------
template <int KDIM, int EPI>
__global__ __launch_bounds__(256) void gemm_kernel(
    const h16* __restrict__ A, int lda, const h16* __restrict__ W,
    const float* __restrict__ bias,
    float* __restrict__ out32, h16* __restrict__ out16, int ldo16)
{
  int lane = threadIdx.x & 31;
  int wave = threadIdx.x >> 5;     // N tile 0..7
  int mtile = blockIdx.x;
  const h16* Ab = A + (size_t)mtile * 16 * lda;

  v8f acc = {};
#pragma unroll
  for (int kt = 0; kt < KDIM / 32; ++kt) {
    v16h a = load_a_frag(Ab, lda, kt * 32, lane);
    v16h b = load_b_frag(W, kt, wave, lane);
    acc = __builtin_amdgcn_wmma_f32_16x16x32_f16(
        false, a, false, b, (short)0, acc, false, false);
  }

  int n = (wave << 4) + (lane & 15);
  float bv = bias[n];
  int mbase = mtile * 16 + ((lane >> 4) << 3);
#pragma unroll
  for (int r = 0; r < 8; ++r) {
    float v = acc[r] + bv;
    if (EPI == 0) v = tanhf(v);
    if (EPI == 2) v = fmaxf(v, 0.0f);
    int row = mbase + r;
    if (EPI == 0 || EPI == 1 || EPI == 2)
      out16[(size_t)row * ldo16 + n] = (h16)v;
    if (EPI == 1 || EPI == 3)
      out32[(size_t)row * 128 + n] = v;
  }
}

// ---------------------------------------------------------------------------
// Softmax over D=128 + gated fusion.  One wave per row, 4 cols per lane.
// ---------------------------------------------------------------------------
__global__ __launch_bounds__(256) void fuse_kernel(
    const float* __restrict__ logits, const float* __restrict__ Zedge,
    const float* __restrict__ Zm, h16* __restrict__ Zf)
{
  int row  = blockIdx.x * 8 + (threadIdx.x >> 5);
  int lane = threadIdx.x & 31;
  float4 g = ((const float4*)(logits + (size_t)row * 128))[lane];

  float m = fmaxf(fmaxf(g.x, g.y), fmaxf(g.z, g.w));
#pragma unroll
  for (int o = 16; o > 0; o >>= 1) m = fmaxf(m, __shfl_xor(m, o, 32));

  float ex = expf(g.x - m), ey = expf(g.y - m);
  float ez = expf(g.z - m), ew = expf(g.w - m);
  float s = ex + ey + ez + ew;
#pragma unroll
  for (int o = 16; o > 0; o >>= 1) s += __shfl_xor(s, o, 32);
  float inv = 1.0f / s;

  float4 ze = ((const float4*)(Zedge + (size_t)row * 128))[lane];
  float4 zm = ((const float4*)(Zm    + (size_t)row * 128))[lane];
  float bx = ex * inv, by = ey * inv, bz = ez * inv, bw = ew * inv;
  *(v4h*)(Zf + (size_t)row * 128 + lane * 4) =
      f4_to_h4(bx * ze.x + (1.f - bx) * zm.x,
               by * ze.y + (1.f - by) * zm.y,
               bz * ze.z + (1.f - bz) * zm.z,
               bw * ze.w + (1.f - bw) * zm.w);
}

// ---------------------------------------------------------------------------
extern "C" void kernel_launch(void* const* d_in, const int* in_sizes, int n_in,
                              void* d_out, int out_size, void* d_ws, size_t ws_size,
                              hipStream_t stream) {
  (void)in_sizes; (void)n_in; (void)out_size; (void)ws_size;
  const int*   batch_nodes = (const int*)d_in[0];
  const float* features    = (const float*)d_in[1];
  const int*   nidx0       = (const int*)d_in[2];
  const int*   nidx1       = (const int*)d_in[3];
  const float* motif_cache = (const float*)d_in[4];
  const float* proj_w1 = (const float*)d_in[5];
  const float* proj_b1 = (const float*)d_in[6];
  const float* proj_w2 = (const float*)d_in[7];
  const float* proj_b2 = (const float*)d_in[8];
  const float* fc_w    = (const float*)d_in[9];
  const float* fc_b    = (const float*)d_in[10];
  const float* gate_w  = (const float*)d_in[11];
  const float* gate_b  = (const float*)d_in[12];
  const float* ffn_w1  = (const float*)d_in[13];
  const float* ffn_b1  = (const float*)d_in[14];
  const float* ffn_w2  = (const float*)d_in[15];
  const float* ffn_b2  = (const float*)d_in[16];

  // workspace layout (bytes)
  char* ws = (char*)d_ws;
  h16* Wsw = (h16*)ws;                              // 1152*128 f16 = 288 KB
  size_t off = (size_t)1152 * 128 * 2;
  h16*   Xc     = (h16*)(ws + off);   off += (size_t)B_ROWS * 384 * 2;  // 48 MB
  h16*   Xfc    = (h16*)(ws + off);   off += (size_t)B_ROWS * 256 * 2;  // 32 MB
  float* Zmotif = (float*)(ws + off); off += (size_t)B_ROWS * 128 * 4;  // 32 MB
  float* Zedge  = (float*)(ws + off); off += (size_t)B_ROWS * 128 * 4;  // 32 MB
  h16*   hbuf   = (h16*)(ws + off);   off += (size_t)B_ROWS * 128 * 2;  // 16 MB
  // dead-buffer reuse:
  float* logits = (float*)Xc;                                   // Xc dead after GEMM1
  h16*   Zfused = (h16*)((char*)Xc + (size_t)B_ROWS * 128 * 4); // Xc[32MB..48MB)
  h16*   tbuf   = (h16*)Zmotif;                                 // motif dead after fuse

  // weight convert + swizzle (offsets in f16 elements: cum K*128)
  swz_kernel<<<(384 * 128 + 255) / 256, 256, 0, stream>>>(proj_w1, Wsw + 0,      384);
  swz_kernel<<<(128 * 128 + 255) / 256, 256, 0, stream>>>(proj_w2, Wsw + 49152,  128);
  swz_kernel<<<(256 * 128 + 255) / 256, 256, 0, stream>>>(fc_w,    Wsw + 65536,  256);
  swz_kernel<<<(128 * 128 + 255) / 256, 256, 0, stream>>>(gate_w,  Wsw + 98304,  128);
  swz_kernel<<<(128 * 128 + 255) / 256, 256, 0, stream>>>(ffn_w1,  Wsw + 114688, 128);
  swz_kernel<<<(128 * 128 + 255) / 256, 256, 0, stream>>>(ffn_w2,  Wsw + 131072, 128);

  // gather + neighbor means (memory-bound phase)
  agg_kernel<<<B_ROWS / 8, 256, 0, stream>>>(batch_nodes, features, nidx0, nidx1,
                                             motif_cache, Xc, Xfc, Zmotif);

  const int MT = B_ROWS / 16;
  // h = tanh(Xc @ W1 + b1)
  gemm_kernel<384, 0><<<MT, 256, 0, stream>>>(Xc, 384, Wsw + 0, proj_b1,
                                              nullptr, hbuf, 128);
  // Z_edge = h @ W2 + b2  (f32 for fusion, f16 into Xfc cols 0..127)
  gemm_kernel<128, 1><<<MT, 256, 0, stream>>>(hbuf, 128, Wsw + 49152, proj_b2,
                                              Zedge, Xfc, 256);
  // hf = relu(Xfc @ fc_w + fc_b)
  gemm_kernel<256, 2><<<MT, 256, 0, stream>>>(Xfc, 256, Wsw + 65536, fc_b,
                                              nullptr, hbuf, 128);
  // logits = hf @ gate_w + gate_b
  gemm_kernel<128, 3><<<MT, 256, 0, stream>>>(hbuf, 128, Wsw + 98304, gate_b,
                                              logits, nullptr, 0);
  // beta = softmax(logits); Z_fused = beta*Z_edge + (1-beta)*Z_motif
  fuse_kernel<<<B_ROWS / 8, 256, 0, stream>>>(logits, Zedge, Zmotif, Zfused);
  // t = relu(Z_fused @ ffn_w1 + ffn_b1)
  gemm_kernel<128, 2><<<MT, 256, 0, stream>>>(Zfused, 128, Wsw + 114688, ffn_b1,
                                              nullptr, tbuf, 128);
  // out = t @ ffn_w2 + ffn_b2  (f32)
  gemm_kernel<128, 3><<<MT, 256, 0, stream>>>(tbuf, 128, Wsw + 131072, ffn_b2,
                                              (float*)d_out, nullptr, 0);
}
